// TULIP_73976516706908
// MI455X (gfx1250) — matrix operand order repaired and checked by reading, not compiled, not run on hardware
//
#include <hip/hip_runtime.h>
#include <hip/hip_bf16.h>
#include <math.h>

// ---------------------------------------------------------------------------
// Problem constants (from reference setup_inputs)
// ---------------------------------------------------------------------------
constexpr int kB   = 2;
constexpr int kM   = 16;
constexpr int kN   = 1024;
constexpr int kC   = 96;          // channels
constexpr int kL   = kM * kN;     // 16384 tokens per batch
constexpr int kT   = kB * kL;     // 32768 tokens total
constexpr int kDI  = 192;         // d_inner
constexpr int kDS  = 16;          // d_state
constexpr int kDTR = 6;           // dt_rank
constexpr int kNX  = 38;          // dtr + 2*ds
constexpr int kNXP = 48;          // padded to multiple of 16
constexpr int kNIN = 384;         // 2*d_inner
constexpr int kNCH = 128;         // scan chunks
constexpr int kCH  = 128;         // chunk length (kNCH*kCH == kL)
constexpr int kKMAX = 192;        // max K over all GEMMs

typedef __attribute__((ext_vector_type(16))) _Float16 v16h;
typedef __attribute__((ext_vector_type(8)))  _Float16 v8h;
typedef __attribute__((ext_vector_type(8)))  float    v8f;

typedef __attribute__((address_space(3))) const _Float16* lds_chptr;
typedef __attribute__((address_space(3))) const v16h*     lds_cv16h;

// ---------------------------------------------------------------------------
// Small math helpers
// ---------------------------------------------------------------------------
__device__ __forceinline__ float silu_f(float v)  { return v / (1.f + __expf(-v)); }
__device__ __forceinline__ float gelu_f(float v)  { return 0.5f * v * (1.f + erff(v * 0.70710678118654752f)); }
__device__ __forceinline__ float softplus_f(float v) { return (v > 20.f) ? v : log1pf(expf(v)); }

// ---------------------------------------------------------------------------
// Weight casting kernels (f32 -> f16, once per launch; cheap)
// ---------------------------------------------------------------------------
__global__ void cast_f16_kernel(const float* __restrict__ src, _Float16* __restrict__ dst, int n) {
    int i = blockIdx.x * blockDim.x + threadIdx.x;
    if (i < n) dst[i] = (_Float16)src[i];
}

// x_proj_w (38,192) -> padded f16 (48,192), rows 38..47 zero
__global__ void pad_xproj_kernel(const float* __restrict__ src, _Float16* __restrict__ dst) {
    int i = blockIdx.x * blockDim.x + threadIdx.x;
    if (i >= kNXP * kDI) return;
    int r = i / kDI, c = i % kDI;
    dst[i] = (r < kNX) ? (_Float16)src[r * kDI + c] : (_Float16)0.f;
}

// ---------------------------------------------------------------------------
// Fused conv1x3 (both directions) + LayerNorm.
// One block = 96 threads (one per out-channel) handling 8 consecutive tokens
// along N (weight values reused across the 8 tokens to cut L2 traffic 8x).
// ---------------------------------------------------------------------------
constexpr int kTOK = 8;
__global__ void conv_ln_kernel(const float* __restrict__ x,
                               const float* __restrict__ w1, const float* __restrict__ b1,
                               const float* __restrict__ w2, const float* __restrict__ b2,
                               const float* __restrict__ lng, const float* __restrict__ lnb,
                               float* __restrict__ t1f, float* __restrict__ t2f,
                               _Float16* __restrict__ t1h, _Float16* __restrict__ t2h) {
    __shared__ float xs[kC][kTOK + 2];
    __shared__ float vals[2][kTOK][kC];
    __shared__ float mv[2][kTOK][2];

    const int t0 = blockIdx.x * kTOK;           // first global token of this block
    const int n0 = t0 % kN;                     // position along N (blocks never cross rows)
    const int c  = threadIdx.x;                 // out-channel

    #pragma unroll
    for (int p = 0; p < kTOK + 2; ++p) {
        int n = n0 + p - 1;
        float v = 0.f;
        if (n >= 0 && n < kN) v = x[(size_t)(t0 + p - 1) * kC + c];
        xs[c][p] = v;
    }
    __syncthreads();

    float s1[kTOK], s2[kTOK];
    #pragma unroll
    for (int tk = 0; tk < kTOK; ++tk) { s1[tk] = b1[c]; s2[tk] = b2[c]; }

    const float* w1r = w1 + c * kC * 3;
    const float* w2r = w2 + c * kC * 3;
    for (int ci = 0; ci < kC; ++ci) {
        float a0 = w1r[ci * 3 + 0], a1 = w1r[ci * 3 + 1], a2 = w1r[ci * 3 + 2];
        float d0 = w2r[ci * 3 + 0], d1 = w2r[ci * 3 + 1], d2 = w2r[ci * 3 + 2];
        #pragma unroll
        for (int tk = 0; tk < kTOK; ++tk) {
            float xm = xs[ci][tk], x0 = xs[ci][tk + 1], xp = xs[ci][tk + 2];
            s1[tk] += a0 * xm + a1 * x0 + a2 * xp;   // forward conv
            s2[tk] += d2 * xm + d1 * x0 + d0 * xp;   // reversed-conv == flipped taps
        }
    }

    #pragma unroll
    for (int tk = 0; tk < kTOK; ++tk) { vals[0][tk][c] = s1[tk]; vals[1][tk][c] = s2[tk]; }
    __syncthreads();

    if (threadIdx.x < 2 * kTOK) {
        int which = threadIdx.x / kTOK, tk = threadIdx.x % kTOK;
        float sum = 0.f, sq = 0.f;
        for (int i = 0; i < kC; ++i) { float v = vals[which][tk][i]; sum += v; sq += v * v; }
        float mu  = sum / kC;
        float var = sq / kC - mu * mu;
        mv[which][tk][0] = mu;
        mv[which][tk][1] = rsqrtf(var + 1e-5f);
    }
    __syncthreads();

    #pragma unroll
    for (int tk = 0; tk < kTOK; ++tk) {
        size_t idx = (size_t)(t0 + tk) * kC + c;
        float v1 = (s1[tk] - mv[0][tk][0]) * mv[0][tk][1] * lng[c] + lnb[c];
        float v2 = (s2[tk] - mv[1][tk][0]) * mv[1][tk][1] * lng[c] + lnb[c];
        t1f[idx] = v1; t2f[idx] = v2;
        t1h[idx] = (_Float16)v1; t2h[idx] = (_Float16)v2;
    }
}

// Generic LayerNorm (f32 in -> f16 out) for the post-SSM norm
__global__ void ln_kernel(const float* __restrict__ in, _Float16* __restrict__ outh,
                          const float* __restrict__ g, const float* __restrict__ bb) {
    __shared__ float vals[kC];
    __shared__ float mv[2];
    int t = blockIdx.x, c = threadIdx.x;
    float v = in[(size_t)t * kC + c];
    vals[c] = v;
    __syncthreads();
    if (c == 0) {
        float s = 0.f, q = 0.f;
        for (int i = 0; i < kC; ++i) { s += vals[i]; q += vals[i] * vals[i]; }
        float mu = s / kC;
        mv[0] = mu;
        mv[1] = rsqrtf(q / kC - mu * mu + 1e-5f);
    }
    __syncthreads();
    outh[(size_t)t * kC + c] = (_Float16)((v - mv[0]) * mv[1] * g[c] + bb[c]);
}

// out_ssm = t1f + t2f
__global__ void add2_kernel(const float* __restrict__ a, const float* __restrict__ b,
                            float* __restrict__ y, int n) {
    int i = blockIdx.x * blockDim.x + threadIdx.x;
    if (i < n) y[i] = a[i] + b[i];
}

// ---------------------------------------------------------------------------
// WMMA GEMM:  Y[T,N] = act( Xh[T,K] @ Wh[N,K]^T + bias ) (+res / +=)
// f16 inputs, f32 accumulate via v_wmma_f32_16x16x32_f16.
//
// Block = 8 waves sharing ONE 16-column B tile. The B tile (16 x K f16,
// <= 6 KB) is staged global->LDS with CDNA5 async-LDS copies
// (global_load_async_to_lds_b128, ASYNCcnt) so the weight stream is read
// from L2 once per 128 output rows instead of once per 16. A fragments are
// streamed from global per wave (no reuse across waves).
//
// Fragment layouts used here (contiguous k order, not the ISA table order):
//   A: lane(l): row = l&15, halves j   -> K = k0+16*(l>>4)+j   (contiguous 32B)
//   B: lane(l): col = l&15, halves j   -> K = k0+16*(l>>4)+j   (contiguous 32B)
//   C/D: elem r of lane l -> row r+8*(l>>4), col l&15
// ---------------------------------------------------------------------------
__device__ __forceinline__ v16h load_a_frag(const _Float16* base, int hi) {
    const v8h lo = *(const v8h*)(base + 8 * hi);
    const v8h hv = *(const v8h*)(base + 16 + 8 * hi);
    v16h r;
    #pragma unroll
    for (int i = 0; i < 8; ++i) { r[i] = lo[i]; r[i + 8] = hv[i]; }
    return r;
}

__global__ void gemm_wmma_kernel(const _Float16* __restrict__ Xh, const _Float16* __restrict__ Wh,
                                 const float* __restrict__ bias, float* __restrict__ Y,
                                 _Float16* __restrict__ Yh, const float* __restrict__ res,
                                 int Tm, int Kk, int Nn, int act_gelu, int accum) {
    __shared__ alignas(64) _Float16 Bs[16 * kKMAX];

    const int ntn  = Nn >> 4;
    const int rblk = blockIdx.x / ntn;           // block of 8 row-tiles
    const int ct   = blockIdx.x % ntn;           // column tile
    const int col0 = ct << 4;
    const int wave = threadIdx.x >> 5;
    const int lane = threadIdx.x & 31;

    // ---- async-stage B tile (16 cols x K halves) into LDS --------------
    {
        lds_chptr bsL = (lds_chptr)&Bs[0];
        const int chunks_per_col = Kk >> 3;          // 16-byte chunks per column
        const int nchunks = 16 * chunks_per_col;     // <= 384
        for (int i = threadIdx.x; i < nchunks; i += blockDim.x) {
            int colr = i / chunks_per_col;
            int kc   = (i % chunks_per_col) << 3;    // half-index within column
            unsigned long long ga =
                (unsigned long long)(uintptr_t)(Wh + (size_t)(col0 + colr) * Kk + kc);
            unsigned int la = (unsigned int)(uintptr_t)(bsL + colr * Kk + kc);
            asm volatile("global_load_async_to_lds_b128 %0, %1, off"
                         :: "v"(la), "v"(ga) : "memory");
        }
        asm volatile("s_wait_asynccnt 0" ::: "memory");
        __syncthreads();
    }

    const int row0 = (rblk * 8 + wave) << 4;
    if (row0 >= Tm) return;

    const int m  = lane & 15;
    const int hi = lane >> 4;

    const _Float16* arow = Xh + (size_t)(row0 + m) * Kk;
    __builtin_prefetch(arow, 0, 1);              // global_prefetch_b8: warm A stream
    lds_chptr brow = (lds_chptr)&Bs[0] + (size_t)m * Kk;

    v8f c = {};
    for (int k0 = 0; k0 < Kk; k0 += 32) {
        v16h a = *(const v16h*)(arow + k0 + 16 * hi);
        v16h b = *(lds_cv16h)(brow + k0 + 16 * hi);   // ds_load from staged tile
        c = __builtin_amdgcn_wmma_f32_16x16x32_f16(
                /*neg_a=*/false, a, /*neg_b=*/false, b,
                /*c_mod=*/(short)0, c, /*reuse_a=*/false, /*reuse_b=*/false);
    }

    const int col = col0 + m;
    const float bval = bias ? bias[col] : 0.f;
    #pragma unroll
    for (int r = 0; r < 8; ++r) {
        int row = row0 + r + 8 * hi;
        float v = c[r] + bval;
        if (act_gelu) v = gelu_f(v);
        size_t idx = (size_t)row * Nn + col;
        if (res) v += res[idx];
        if (accum) Y[idx] += v; else Y[idx] = v;
        if (Yh) Yh[idx] = (_Float16)v;
    }
}

// ---------------------------------------------------------------------------
// Depthwise causal conv1d (k=4) over L + bias + SiLU. Reads xm slice of xz.
// ---------------------------------------------------------------------------
__global__ void dwconv_silu_kernel(const float* __restrict__ xz, const float* __restrict__ w,
                                   const float* __restrict__ bias, float* __restrict__ xc,
                                   _Float16* __restrict__ xch) {
    size_t i = (size_t)blockIdx.x * blockDim.x + threadIdx.x;
    if (i >= (size_t)kT * kDI) return;
    int d = (int)(i % kDI);
    size_t t = i / kDI;
    int l = (int)(t % kL);
    float s = bias[d];
    #pragma unroll
    for (int j = 0; j < 4; ++j) {
        int ll = l - 3 + j;
        if (ll >= 0) s += w[d * 4 + j] * xz[(t - 3 + j) * kNIN + d];
    }
    s = silu_f(s);
    xc[i] = s;
    xch[i] = (_Float16)s;
}

// dt = softplus(x_dbl[:, :6] @ dt_proj_w^T + dt_proj_b)
__global__ void dt_kernel(const float* __restrict__ xdbl, const float* __restrict__ w,
                          const float* __restrict__ b, float* __restrict__ dtb) {
    size_t i = (size_t)blockIdx.x * blockDim.x + threadIdx.x;
    if (i >= (size_t)kT * kDI) return;
    int d = (int)(i % kDI);
    size_t t = i / kDI;
    const float* xr = xdbl + t * kNXP;
    float s = b[d];
    #pragma unroll
    for (int r = 0; r < kDTR; ++r) s += xr[r] * w[d * kDTR + r];
    dtb[i] = softplus_f(s);
}

// ---------------------------------------------------------------------------
// Chunked parallel selective scan.
// Phase 1: per chunk compute decay product P and zero-init partial state S.
// Phase 2: sequential combine over 128 chunks -> incoming state per chunk.
// Phase 3: re-scan chunk with correct h0, emit g = (y + u*D) * silu(z) in f16.
// Layout of P/S/Hin: [((b*DI + d)*DS + s)*NCH + ch]
// ---------------------------------------------------------------------------
__global__ void scan_p1_kernel(const float* __restrict__ dtb, const float* __restrict__ xc,
                               const float* __restrict__ xdbl, const float* __restrict__ A_log,
                               float* __restrict__ P, float* __restrict__ S) {
    const int d  = threadIdx.x;                 // 0..191
    const int ch = blockIdx.x % kNCH;
    const int b  = blockIdx.x / kNCH;
    float A[kDS], Pp[kDS], H[kDS];
    #pragma unroll
    for (int s = 0; s < kDS; ++s) { A[s] = -expf(A_log[d * kDS + s]); Pp[s] = 1.f; H[s] = 0.f; }
    const size_t base = (size_t)b * kL + (size_t)ch * kCH;
    for (int t = 0; t < kCH; ++t) {
        size_t tt = base + t;
        float dtv = dtb[tt * kDI + d];
        float du  = dtv * xc[tt * kDI + d];
        const float* xr = xdbl + tt * kNXP;
        #pragma unroll
        for (int s = 0; s < kDS; ++s) {
            float da = __expf(dtv * A[s]);
            H[s] = da * H[s] + du * xr[kDTR + s];
            Pp[s] *= da;
        }
    }
    #pragma unroll
    for (int s = 0; s < kDS; ++s) {
        size_t o = (((size_t)b * kDI + d) * kDS + s) * kNCH + ch;
        P[o] = Pp[s]; S[o] = H[s];
    }
}

__global__ void scan_p2_kernel(const float* __restrict__ P, const float* __restrict__ S,
                               float* __restrict__ Hin) {
    int i = blockIdx.x * blockDim.x + threadIdx.x;     // over kB*kDI*kDS = 6144
    if (i >= kB * kDI * kDS) return;
    size_t base = (size_t)i * kNCH;
    float H = 0.f;
    for (int ch = 0; ch < kNCH; ++ch) {
        Hin[base + ch] = H;
        H = P[base + ch] * H + S[base + ch];
    }
}

__global__ void scan_p3_kernel(const float* __restrict__ dtb, const float* __restrict__ xc,
                               const float* __restrict__ xdbl, const float* __restrict__ A_log,
                               const float* __restrict__ Dp, const float* __restrict__ xz,
                               const float* __restrict__ Hin, _Float16* __restrict__ gh) {
    const int d  = threadIdx.x;
    const int ch = blockIdx.x % kNCH;
    const int b  = blockIdx.x / kNCH;
    float A[kDS], H[kDS];
    #pragma unroll
    for (int s = 0; s < kDS; ++s) {
        A[s] = -expf(A_log[d * kDS + s]);
        H[s] = Hin[(((size_t)b * kDI + d) * kDS + s) * kNCH + ch];
    }
    const float dval = Dp[d];
    const size_t base = (size_t)b * kL + (size_t)ch * kCH;
    for (int t = 0; t < kCH; ++t) {
        size_t tt = base + t;
        float dtv = dtb[tt * kDI + d];
        float u   = xc[tt * kDI + d];
        float du  = dtv * u;
        const float* xr = xdbl + tt * kNXP;
        float y = 0.f;
        #pragma unroll
        for (int s = 0; s < kDS; ++s) {
            float da = __expf(dtv * A[s]);
            H[s] = da * H[s] + du * xr[kDTR + s];
            y += H[s] * xr[kDTR + kDS + s];
        }
        y += u * dval;
        float z = xz[tt * kNIN + kDI + d];
        gh[tt * kDI + d] = (_Float16)(y * silu_f(z));
    }
}

// ---------------------------------------------------------------------------
// Host-side orchestration
// ---------------------------------------------------------------------------
extern "C" void kernel_launch(void* const* d_in, const int* in_sizes, int n_in,
                              void* d_out, int out_size, void* d_ws, size_t ws_size,
                              hipStream_t stream) {
    (void)in_sizes; (void)n_in; (void)out_size; (void)ws_size;

    const float* x          = (const float*)d_in[0];
    const float* conv1_w    = (const float*)d_in[1];
    const float* conv1_b    = (const float*)d_in[2];
    const float* conv2_w    = (const float*)d_in[3];
    const float* conv2_b    = (const float*)d_in[4];
    const float* ln_g       = (const float*)d_in[5];
    const float* ln_b       = (const float*)d_in[6];
    const float* in_proj_w  = (const float*)d_in[7];
    const float* conv1d_w   = (const float*)d_in[8];
    const float* conv1d_b   = (const float*)d_in[9];
    const float* x_proj_w   = (const float*)d_in[10];
    const float* dt_proj_w  = (const float*)d_in[11];
    const float* dt_proj_b  = (const float*)d_in[12];
    const float* A_log      = (const float*)d_in[13];
    const float* Dp         = (const float*)d_in[14];
    const float* out_proj_w = (const float*)d_in[15];
    const float* fc1_w      = (const float*)d_in[16];
    const float* fc1_b      = (const float*)d_in[17];
    const float* fc2_w      = (const float*)d_in[18];
    const float* fc2_b      = (const float*)d_in[19];
    float* out = (float*)d_out;

    // Deterministic bump allocator over d_ws (256B aligned).
    char* ws = (char*)d_ws;
    size_t off = 0;
    auto alloc = [&](size_t bytes) -> char* {
        char* p = ws + off;
        off = (off + bytes + 255) & ~(size_t)255;
        return p;
    };

    _Float16* whIn  = (_Float16*)alloc((size_t)kNIN * kC  * 2);   // in_proj  (384,96)
    _Float16* whXp  = (_Float16*)alloc((size_t)kNXP * kDI * 2);   // x_proj padded (48,192)
    _Float16* whOut = (_Float16*)alloc((size_t)kC   * kDI * 2);   // out_proj (96,192)
    _Float16* whF1  = (_Float16*)alloc((size_t)kC   * kC  * 2);   // fc1 (96,96)
    _Float16* whF2  = (_Float16*)alloc((size_t)kC   * kC  * 2);   // fc2 (96,96)

    float*    t1f   = (float*)   alloc((size_t)kT * kC  * 4);
    float*    t2f   = (float*)   alloc((size_t)kT * kC  * 4);
    _Float16* t1h   = (_Float16*)alloc((size_t)kT * kC  * 2);
    _Float16* t2h   = (_Float16*)alloc((size_t)kT * kC  * 2);
    float*    ossm  = (float*)   alloc((size_t)kT * kC  * 4);
    float*    xz    = (float*)   alloc((size_t)kT * kNIN * 4);
    float*    xcf   = (float*)   alloc((size_t)kT * kDI * 4);
    _Float16* xch   = (_Float16*)alloc((size_t)kT * kDI * 2);
    float*    xdbl  = (float*)   alloc((size_t)kT * kNXP * 4);
    float*    dtb   = (float*)   alloc((size_t)kT * kDI * 4);
    float*    Pb    = (float*)   alloc((size_t)kB * kDI * kDS * kNCH * 4);
    float*    Sb    = (float*)   alloc((size_t)kB * kDI * kDS * kNCH * 4);
    float*    Hin   = (float*)   alloc((size_t)kB * kDI * kDS * kNCH * 4);
    _Float16* gh    = (_Float16*)alloc((size_t)kT * kDI * 2);
    _Float16* lnh   = (_Float16*)alloc((size_t)kT * kC  * 2);
    float*    fc1f  = (float*)   alloc((size_t)kT * kC  * 4);
    _Float16* g1h   = (_Float16*)alloc((size_t)kT * kC  * 2);

    auto cdiv = [](int a, int b) { return (a + b - 1) / b; };

    // --- weights -> f16 -------------------------------------------------
    cast_f16_kernel<<<cdiv(kNIN * kC, 256), 256, 0, stream>>>(in_proj_w,  whIn,  kNIN * kC);
    pad_xproj_kernel<<<cdiv(kNXP * kDI, 256), 256, 0, stream>>>(x_proj_w, whXp);
    cast_f16_kernel<<<cdiv(kC * kDI, 256), 256, 0, stream>>>(out_proj_w, whOut, kC * kDI);
    cast_f16_kernel<<<cdiv(kC * kC, 256), 256, 0, stream>>>(fc1_w, whF1, kC * kC);
    cast_f16_kernel<<<cdiv(kC * kC, 256), 256, 0, stream>>>(fc2_w, whF2, kC * kC);

    // --- conv1x3 (both dirs) + LN --------------------------------------
    conv_ln_kernel<<<kT / kTOK, kC, 0, stream>>>(x, conv1_w, conv1_b, conv2_w, conv2_b,
                                                 ln_g, ln_b, t1f, t2f, t1h, t2h);

    // --- out_ssm = t1 + t2 ----------------------------------------------
    add2_kernel<<<cdiv(kT * kC, 256), 256, 0, stream>>>(t1f, t2f, ossm, kT * kC);

    auto launch_gemm = [&](const _Float16* Xh, const _Float16* Wh, const float* bias,
                           float* Y, _Float16* Yh, const float* res,
                           int Tm, int Kk, int Nn, int act, int acc) {
        int blocks = (Tm / 128) * (Nn / 16);   // 8 row-tiles per block share one B tile
        gemm_wmma_kernel<<<blocks, 256, 0, stream>>>(Xh, Wh, bias, Y, Yh, res,
                                                     Tm, Kk, Nn, act, acc);
    };

    // --- two Mamba passes (shared params, different inputs) -------------
    for (int dir = 0; dir < 2; ++dir) {
        const _Float16* th = dir ? t2h : t1h;

        // xz = t @ in_proj^T           (32768 x 96 x 384)
        launch_gemm(th, whIn, nullptr, xz, nullptr, nullptr, kT, kC, kNIN, 0, 0);

        // depthwise causal conv + SiLU
        dwconv_silu_kernel<<<cdiv(kT * kDI, 256), 256, 0, stream>>>(xz, conv1d_w, conv1d_b, xcf, xch);

        // x_dbl = xc @ x_proj^T (padded N=48)   (32768 x 192 x 48)
        launch_gemm(xch, whXp, nullptr, xdbl, nullptr, nullptr, kT, kDI, kNXP, 0, 0);

        // dt = softplus(x_dbl[:, :6] @ dt_proj^T + b)
        dt_kernel<<<cdiv(kT * kDI, 256), 256, 0, stream>>>(xdbl, dt_proj_w, dt_proj_b, dtb);

        // chunked parallel selective scan
        scan_p1_kernel<<<kB * kNCH, kDI, 0, stream>>>(dtb, xcf, xdbl, A_log, Pb, Sb);
        scan_p2_kernel<<<cdiv(kB * kDI * kDS, 256), 256, 0, stream>>>(Pb, Sb, Hin);
        scan_p3_kernel<<<kB * kNCH, kDI, 0, stream>>>(dtb, xcf, xdbl, A_log, Dp, xz, Hin, gh);

        // out_ssm += (y * silu(z)) @ out_proj^T   (32768 x 192 x 96)
        launch_gemm(gh, whOut, nullptr, ossm, nullptr, nullptr, kT, kDI, kC, 0, 1);
    }

    // --- final LN + MLP (+residual) -------------------------------------
    ln_kernel<<<kT, kC, 0, stream>>>(ossm, lnh, ln_g, ln_b);

    // fc1: gelu(lnh @ fc1^T + b1), keep f16 copy for fc2
    launch_gemm(lnh, whF1, fc1_b, fc1f, g1h, nullptr, kT, kC, kC, 1, 0);

    // fc2: out = g1 @ fc2^T + b2 + out_ssm  (writes every element of d_out)
    launch_gemm(g1h, whF2, fc2_b, out, nullptr, ossm, kT, kC, kC, 0, 0);
}
